// MultiheadSelfAttention_84859963834760
// MI455X (gfx1250) — matrix operand order repaired and checked by reading, not compiled, not run on hardware
//
#include <hip/hip_runtime.h>

typedef __attribute__((ext_vector_type(16))) __bf16        v16bf;
typedef __attribute__((ext_vector_type(8)))  float         v8f;
typedef __attribute__((ext_vector_type(8)))  unsigned int  v8u;

#define DM    1024
#define HEADS 16
#define DK    64
#define SEQ   2048
#define BATCH 4
#define MTOT  (BATCH * SEQ)

// ---------- helpers ----------

static __device__ __forceinline__ unsigned short f2bf(float f) {
  unsigned u = __float_as_uint(f);
  u += 0x7FFFu + ((u >> 16) & 1u);          // round-to-nearest-even
  return (unsigned short)(u >> 16);
}

static __device__ __forceinline__ unsigned long long pack4bf(float a, float b,
                                                             float c, float d) {
  return (unsigned long long)f2bf(a) | ((unsigned long long)f2bf(b) << 16) |
         ((unsigned long long)f2bf(c) << 32) | ((unsigned long long)f2bf(d) << 48);
}

static __device__ __forceinline__ v8f zero8() {
  v8f z;
#pragma unroll
  for (int i = 0; i < 8; ++i) z[i] = 0.0f;
  return z;
}

// A-matrix 16x32 bf16 fragment (this code's contiguous k order): lane L: m=L&15, half=L>>4;
// VGPR v holds K pair kb,kb+1 with kb = 16*half + 2v.
static __device__ __forceinline__ v16bf load_a_frag(const unsigned short* base,
                                                    int ld, int m, int koff,
                                                    int half) {
  v8u r;
#pragma unroll
  for (int v = 0; v < 8; ++v) {
    int kb = 16 * half + 2 * v;
    r[v] = *(const unsigned int*)(base + (size_t)m * ld + koff + kb);
  }
  return __builtin_bit_cast(v16bf, r);
}

// B-matrix 32x16 bf16 fragment: lane L: n=L&15, half=L>>4; VGPR v holds
// K pair (16*half + 2v, +1).  Source stored "B^T": element (k,n) at base[n*ld+k].
static __device__ __forceinline__ v16bf load_b_frag(const unsigned short* base,
                                                    int ld, int n, int koff,
                                                    int half) {
  v8u r;
#pragma unroll
  for (int v = 0; v < 8; ++v) {
    int kk = 16 * half + 2 * v;
    r[v] = *(const unsigned int*)(base + (size_t)n * ld + koff + kk);
  }
  return __builtin_bit_cast(v16bf, r);
}

static __device__ __forceinline__ v8f wmma_bf16(v16bf a, v16bf b, v8f c) {
  return __builtin_amdgcn_wmma_f32_16x16x32_bf16(false, a, false, b, (short)0,
                                                 c, false, false);
}

// ---------- kernel 1: QKV projection (+RoPE for Q/K, transpose for V) ----------
// out = x @ W^T.  WG tile 128(M) x 128(N), 8 waves as 4x2, wave tile 32x64
// (2x4 frags), K-chunk 64.  MODE 0: RoPE -> bf16 head-major [bh][pos][dk];
// MODE 1: plain -> bf16 transposed [bh][dk][pos].
template <int MODE>
__global__ __launch_bounds__(256) void qkv_gemm_kernel(
    const float* __restrict__ x, const float* __restrict__ W,
    unsigned short* __restrict__ out) {
  __shared__ __align__(16) unsigned short lA[128 * 64];  // x tile bf16 (16KB)
  __shared__ __align__(16) unsigned short lB[128 * 64];  // W tile bf16 [n][k]
  const int tid = threadIdx.x;
  const int wave = tid >> 5, lane = tid & 31;
  const int half = lane >> 4, mr = lane & 15;
  const int bm = blockIdx.x;   // 128-row tile
  const int bn = blockIdx.y;   // 128-col tile (2 heads)
  const int wm = (wave >> 1) * 32, wn = (wave & 1) * 64;

  v8f acc[2][4];
#pragma unroll
  for (int i = 0; i < 2; ++i)
#pragma unroll
    for (int j = 0; j < 4; ++j) acc[i][j] = zero8();

  for (int k0 = 0; k0 < DM; k0 += 64) {
    // fill tiles: float4 global reads -> packed u64 LDS stores
    for (int i = tid; i < 128 * 16; i += 256) {
      int r = i >> 4, c4 = (i & 15) * 4;
      float4 f = *(const float4*)&x[(size_t)(bm * 128 + r) * DM + k0 + c4];
      *(unsigned long long*)&lA[r * 64 + c4] = pack4bf(f.x, f.y, f.z, f.w);
    }
    for (int i = tid; i < 128 * 16; i += 256) {
      int r = i >> 4, c4 = (i & 15) * 4;
      float4 f = *(const float4*)&W[(size_t)(bn * 128 + r) * DM + k0 + c4];
      *(unsigned long long*)&lB[r * 64 + c4] = pack4bf(f.x, f.y, f.z, f.w);
    }
    if (k0 + 64 < DM)
      __builtin_prefetch(&x[(size_t)(bm * 128 + (tid >> 1)) * DM + k0 + 64], 0, 1);
    __syncthreads();
#pragma unroll
    for (int c = 0; c < 2; ++c) {  // two 32-wide K steps
      v16bf a[2];
#pragma unroll
      for (int i = 0; i < 2; ++i)
        a[i] = load_a_frag(lA, 64, wm + i * 16 + mr, c * 32, half);
#pragma unroll
      for (int j = 0; j < 4; ++j) {
        v16bf b = load_b_frag(lB, 64, wn + j * 16 + mr, c * 32, half);
#pragma unroll
        for (int i = 0; i < 2; ++i) acc[i][j] = wmma_bf16(a[i], b, acc[i][j]);
      }
    }
    __syncthreads();
  }

#pragma unroll
  for (int i = 0; i < 2; ++i) {
#pragma unroll
    for (int j = 0; j < 4; ++j) {
      const int n_local = j * 16 + mr;                 // 0..63 within head
      const int h = bn * 2 + (wn >> 6);                // head index
      if (MODE == 0) {
        // inv_freq = theta^(-2t/64) = exp(-t * ln(1e4)/32)
        const float invf = __expf(-(float)(n_local >> 1) * 0.28782387f);
#pragma unroll
        for (int r = 0; r < 8; ++r) {
          int gm = bm * 128 + wm + i * 16 + r + 8 * half;
          int bi = gm >> 11, pos = gm & (SEQ - 1);
          float val = acc[i][j][r];
          float other = __shfl_xor(val, 1, 32);        // RoPE partner (col ^ 1)
          float sn, cs;
          __sincosf((float)pos * invf, &sn, &cs);
          float res = (n_local & 1) ? (other * sn + val * cs)
                                    : (val * cs - other * sn);
          out[((size_t)(bi * HEADS + h) * SEQ + pos) * DK + n_local] = f2bf(res);
        }
      } else {
#pragma unroll
        for (int r = 0; r < 8; ++r) {
          int gm = bm * 128 + wm + i * 16 + r + 8 * half;
          int bi = gm >> 11, pos = gm & (SEQ - 1);
          out[((size_t)(bi * HEADS + h) * DK + n_local) * SEQ + pos] =
              f2bf(acc[i][j][r]);
        }
      }
    }
  }
}

// ---------- kernel 2: causal flash attention ----------
// grid (SEQ/128, BATCH*HEADS); 8 waves/WG; each wave owns 16 query rows.
// No __syncthreads (per-wave causal trip counts differ); LDS is wave-private.
__global__ __launch_bounds__(256) void attn_kernel(
    const unsigned short* __restrict__ Qb, const unsigned short* __restrict__ Kb,
    const unsigned short* __restrict__ Vt, unsigned short* __restrict__ attn) {
  __shared__ __align__(16) unsigned short lP[8][16 * 32];  // per-wave P tile
  const int tid = threadIdx.x;
  const int wave = tid >> 5, lane = tid & 31;
  const int half = lane >> 4, mr = lane & 15;
  const int bh = blockIdx.y;
  const int bi = bh >> 4, h = bh & 15;
  const int q_lo = blockIdx.x * 128 + wave * 16;

  // Q A-fragments (16 x 64 = two 16x32 frags), held in registers
  const unsigned short* Qrow = Qb + ((size_t)bh * SEQ + q_lo) * DK;
  v8u q0r, q1r;
#pragma unroll
  for (int v = 0; v < 8; ++v) {
    int kb = 16 * half + 2 * v;
    q0r[v] = *(const unsigned int*)(Qrow + (size_t)mr * DK + kb);
    q1r[v] = *(const unsigned int*)(Qrow + (size_t)mr * DK + 32 + kb);
  }
  const v16bf q0 = __builtin_bit_cast(v16bf, q0r);
  const v16bf q1 = __builtin_bit_cast(v16bf, q1r);

  v8f o[4];
#pragma unroll
  for (int f = 0; f < 4; ++f) o[f] = zero8();
  float mrow[8], lrow[8];
#pragma unroll
  for (int r = 0; r < 8; ++r) { mrow[r] = -__builtin_inff(); lrow[r] = 0.0f; }

  unsigned short* P = &lP[wave][0];
  const unsigned short* Kbase = Kb + (size_t)bh * SEQ * DK;
  const unsigned short* Vbase = Vt + (size_t)bh * DK * SEQ;

  const int last_kb = (q_lo + 15) >> 5;
  for (int kbI = 0; kbI <= last_kb; ++kbI) {
    const int kbase = kbI * 32;
    // ---- S = Q K^T (16 q-rows x 32 keys) ----
    v8f s[2];
#pragma unroll
    for (int g = 0; g < 2; ++g) {
      const int key = kbase + g * 16 + mr;
      const unsigned short* Krow = Kbase + (size_t)key * DK;
      v8u b0r, b1r;
#pragma unroll
      for (int v = 0; v < 8; ++v) {
        int kk = 16 * half + 2 * v;
        b0r[v] = *(const unsigned int*)(Krow + kk);
        b1r[v] = *(const unsigned int*)(Krow + 32 + kk);
      }
      v8f z = zero8();
      z = wmma_bf16(q0, __builtin_bit_cast(v16bf, b0r), z);
      s[g] = wmma_bf16(q1, __builtin_bit_cast(v16bf, b1r), z);
    }
    // ---- scale + causal mask ----
#pragma unroll
    for (int g = 0; g < 2; ++g) {
      const int key = kbase + g * 16 + mr;
#pragma unroll
      for (int r = 0; r < 8; ++r) {
        int qrow = q_lo + r + 8 * half;
        float sv = s[g][r] * 0.125f;  // 1/sqrt(64)
        s[g][r] = (key > qrow) ? -__builtin_inff() : sv;
      }
    }
    // ---- online softmax (row stats across 16-lane groups) ----
#pragma unroll
    for (int r = 0; r < 8; ++r) {
      float v = fmaxf(s[0][r], s[1][r]);
      v = fmaxf(v, __shfl_xor(v, 1, 32));
      v = fmaxf(v, __shfl_xor(v, 2, 32));
      v = fmaxf(v, __shfl_xor(v, 4, 32));
      v = fmaxf(v, __shfl_xor(v, 8, 32));
      float mnew = fmaxf(mrow[r], v);
      float alpha = __expf(mrow[r] - mnew);
      mrow[r] = mnew;
      float p0 = __expf(s[0][r] - mnew);
      float p1 = __expf(s[1][r] - mnew);
      s[0][r] = p0; s[1][r] = p1;
      float rs = p0 + p1;
      rs += __shfl_xor(rs, 1, 32);
      rs += __shfl_xor(rs, 2, 32);
      rs += __shfl_xor(rs, 4, 32);
      rs += __shfl_xor(rs, 8, 32);
      lrow[r] = lrow[r] * alpha + rs;
#pragma unroll
      for (int f = 0; f < 4; ++f) o[f][r] *= alpha;
    }
    // ---- P: C-layout -> A-layout via wave-private LDS ----
#pragma unroll
    for (int g = 0; g < 2; ++g)
#pragma unroll
      for (int r = 0; r < 8; ++r)
        P[(r + 8 * half) * 32 + g * 16 + mr] = f2bf(s[g][r]);
    __asm__ volatile("s_wait_dscnt 0" ::: "memory");
    v8u praw;
#pragma unroll
    for (int v = 0; v < 8; ++v) {
      int kb = 16 * half + 2 * v;
      praw[v] = *(const unsigned int*)(P + mr * 32 + kb);
    }
    const v16bf pA = __builtin_bit_cast(v16bf, praw);
    // ---- O += P V  (V^T stored [d][pos] -> contiguous key pairs) ----
#pragma unroll
    for (int f = 0; f < 4; ++f) {
      const unsigned short* Vrow = Vbase + (size_t)(f * 16 + mr) * SEQ;
      v8u vraw;
#pragma unroll
      for (int v = 0; v < 8; ++v) {
        int key = kbase + 16 * half + 2 * v;
        vraw[v] = *(const unsigned int*)(Vrow + key);
      }
      o[f] = wmma_bf16(pA, __builtin_bit_cast(v16bf, vraw), o[f]);
    }
  }
  // ---- normalize, write merged-head bf16 [b*s][1024] ----
#pragma unroll
  for (int r = 0; r < 8; ++r) {
    float inv = 1.0f / lrow[r];
    int pos = q_lo + r + 8 * half;
    size_t rowoff = ((size_t)bi * SEQ + pos) * DM + (size_t)h * DK;
#pragma unroll
    for (int f = 0; f < 4; ++f)
      attn[rowoff + f * 16 + mr] = f2bf(o[f][r] * inv);
  }
}

// ---------- kernel 3: output projection  out = attn @ Wo^T (fp32 out) ----------
// Same 128x128 / 2x4-frag tiling; A is already bf16 in workspace.
__global__ __launch_bounds__(256) void out_proj_kernel(
    const unsigned short* __restrict__ A, const float* __restrict__ W,
    float* __restrict__ out) {
  __shared__ __align__(16) unsigned short lA[128 * 64];
  __shared__ __align__(16) unsigned short lB[128 * 64];
  const int tid = threadIdx.x;
  const int wave = tid >> 5, lane = tid & 31;
  const int half = lane >> 4, mr = lane & 15;
  const int bm = blockIdx.x, bn = blockIdx.y;
  const int wm = (wave >> 1) * 32, wn = (wave & 1) * 64;

  v8f acc[2][4];
#pragma unroll
  for (int i = 0; i < 2; ++i)
#pragma unroll
    for (int j = 0; j < 4; ++j) acc[i][j] = zero8();

  for (int k0 = 0; k0 < DM; k0 += 64) {
    for (int i = tid; i < 128 * 16; i += 256) {  // bf16 A: straight u64 copies
      int r = i >> 4, c4 = (i & 15) * 4;
      *(unsigned long long*)&lA[r * 64 + c4] =
          *(const unsigned long long*)&A[(size_t)(bm * 128 + r) * DM + k0 + c4];
    }
    for (int i = tid; i < 128 * 16; i += 256) {
      int r = i >> 4, c4 = (i & 15) * 4;
      float4 f = *(const float4*)&W[(size_t)(bn * 128 + r) * DM + k0 + c4];
      *(unsigned long long*)&lB[r * 64 + c4] = pack4bf(f.x, f.y, f.z, f.w);
    }
    if (k0 + 64 < DM)
      __builtin_prefetch(&A[(size_t)(bm * 128 + (tid >> 1)) * DM + k0 + 64], 0, 1);
    __syncthreads();
#pragma unroll
    for (int c = 0; c < 2; ++c) {
      v16bf a[2];
#pragma unroll
      for (int i = 0; i < 2; ++i)
        a[i] = load_a_frag(lA, 64, wm + i * 16 + mr, c * 32, half);
#pragma unroll
      for (int j = 0; j < 4; ++j) {
        v16bf b = load_b_frag(lB, 64, wn + j * 16 + mr, c * 32, half);
#pragma unroll
        for (int i = 0; i < 2; ++i) acc[i][j] = wmma_bf16(a[i], b, acc[i][j]);
      }
    }
    __syncthreads();
  }
#pragma unroll
  for (int i = 0; i < 2; ++i)
#pragma unroll
    for (int j = 0; j < 4; ++j) {
      int gn = bn * 128 + wn + j * 16 + mr;
#pragma unroll
      for (int r = 0; r < 8; ++r) {
        int gm = bm * 128 + wm + i * 16 + r + 8 * half;
        out[(size_t)gm * DM + gn] = acc[i][j][r];
      }
    }
}

// ---------- host ----------
extern "C" void kernel_launch(void* const* d_in, const int* in_sizes, int n_in,
                              void* d_out, int out_size, void* d_ws,
                              size_t ws_size, hipStream_t stream) {
  const float* x  = (const float*)d_in[0];
  const float* Wq = (const float*)d_in[1];
  const float* Wk = (const float*)d_in[2];
  const float* Wv = (const float*)d_in[3];
  const float* Wo = (const float*)d_in[4];

  const size_t elems = (size_t)MTOT * DM;  // 8Mi elements per buffer
  unsigned short* Qb = (unsigned short*)d_ws;      // [bh][pos][dk] bf16 (roped)
  unsigned short* Kb = Qb + elems;                 // [bh][pos][dk] bf16 (roped)
  unsigned short* Vt = Kb + elems;                 // [bh][dk][pos] bf16
  unsigned short* At = Vt + elems;                 // [b*s][1024]   bf16

  dim3 blk(256);
  dim3 g1(MTOT / 128, DM / 128);
  qkv_gemm_kernel<0><<<g1, blk, 0, stream>>>(x, Wq, Qb);
  qkv_gemm_kernel<0><<<g1, blk, 0, stream>>>(x, Wk, Kb);
  qkv_gemm_kernel<1><<<g1, blk, 0, stream>>>(x, Wv, Vt);

  dim3 g2(SEQ / 128, BATCH * HEADS);
  attn_kernel<<<g2, blk, 0, stream>>>(Qb, Kb, Vt, At);

  dim3 g3(MTOT / 128, DM / 128);
  out_proj_kernel<<<g3, blk, 0, stream>>>(At, Wo, (float*)d_out);
}
